// Democracy_loss_71880572666198
// MI455X (gfx1250) — compile-verified, matrix operand order checked by reading, not run on hardware
//
#include <hip/hip_runtime.h>
#include <stdint.h>

// ---------------- constants from the reference ----------------
#define TEMP    0.07f
#define EPSF    1e-12f
#define IN_DIM  120000          // 300*25*16
#define NF      256             // further-pair samples
#define NC      64              // closest-pair samples
#define NTOT    320             // NF + NC
#define DIM     128             // embedding width
#define MROWS   64              // M rows per gemm1 block (4 sub-tiles of 16)
#define KSPLIT  50              // K-split blocks for GEMM1
#define KCHUNK  (IN_DIM / KSPLIT)   // 2400
#define KSTEPS  (KCHUNK / 32)       // 75 K=32 steps per block

typedef float        v8f   __attribute__((ext_vector_type(8)));
typedef unsigned int v8u   __attribute__((ext_vector_type(8)));
typedef __bf16       v16bf __attribute__((ext_vector_type(16)));

// fp32 -> bf16 with round-to-nearest-even, packed two per dword
__device__ __forceinline__ uint32_t f2bf1(float f) {
    uint32_t u = __builtin_bit_cast(uint32_t, f);
    return (u + 0x7FFFu + ((u >> 16) & 1u)) >> 16;
}
__device__ __forceinline__ uint32_t packbf2(float lo, float hi) {
    return f2bf1(lo) | (f2bf1(hi) << 16);
}

// ---------------- kernel 0: zero workspace accumulators ----------------
__global__ void zero_ws_kernel(float* __restrict__ hacc, float* __restrict__ accum) {
    int i = blockIdx.x * blockDim.x + threadIdx.x;
    if (i < NTOT * DIM) hacc[i] = 0.0f;
    if (i < 2) accum[i] = 0.0f;
}

// ---------------- kernel 1: X @ W1 (bf16 WMMA, fp32 accumulate, K-split + atomics) ----
// grid = (5 M-tiles of 64 rows, KSPLIT); block = 256 (8 wave32 waves).
// Wave w owns N-tile w (16 cols) for all 4 M-subtiles -> 4 WMMAs per K-step
// against a single B fragment (4x matrix work per staged byte vs 16-row tiles).
__global__ __launch_bounds__(256)
void gemm1_kernel(const float* __restrict__ Xf, const float* __restrict__ Xc,
                  const float* __restrict__ W1, float* __restrict__ hacc)
{
    __shared__ uint32_t As[MROWS * 17];   // 64 rows x 16 bf16-pairs (K=32), padded stride
    __shared__ uint32_t Bs[16 * 132];     // 16 K-pairs x 128 cols, padded stride

    const int tid  = threadIdx.x;
    const int lane = tid & 31;
    const int wave = tid >> 5;
    const int row0 = blockIdx.x * MROWS;
    const int kbase = blockIdx.y * KCHUNK;

    // NF == 256 is a multiple of MROWS, so a block never straddles Xf/Xc
    const float* X = (row0 < NF) ? (Xf + (size_t)row0 * IN_DIM)
                                 : (Xc + (size_t)(row0 - NF) * IN_DIM);

    v8f acc0 = {0.f,0.f,0.f,0.f,0.f,0.f,0.f,0.f};
    v8f acc1 = acc0, acc2 = acc0, acc3 = acc0;

    const int m    = lane & 15;
    const int hi   = lane >> 4;
    const int ncol = wave * 16 + m;       // this lane's output column

    for (int kb = 0; kb < KSTEPS; ++kb) {
        const int k0 = kbase + kb * 32;
        __syncthreads();
        #pragma unroll
        for (int i = 0; i < MROWS * 16 / 256; ++i) {   // stage A: 64x32 fp32 -> bf16 pairs
            int idx = tid + i * 256;                   // 0..1023
            int am = idx >> 4, ap = idx & 15;
            const float* s = X + (size_t)am * IN_DIM + k0 + 2 * ap;
            As[am * 17 + ap] = packbf2(s[0], s[1]);
            if (kb + 1 < KSTEPS) __builtin_prefetch(s + 32, 0, 3);  // near-cache prefetch
        }
        #pragma unroll
        for (int i = 0; i < 8; ++i) {                  // stage B: 32x128 fp32 -> bf16 pairs
            int idx = tid + i * 256;                   // 0..2047
            int bp = idx >> 7, bn = idx & 127;
            const float* s = W1 + (size_t)(k0 + 2 * bp) * DIM + bn;
            Bs[bp * 132 + bn] = packbf2(s[0], s[DIM]);
            if (kb + 1 < KSTEPS) __builtin_prefetch(s + 32 * DIM, 0, 3);
        }
        __syncthreads();

        // B fragment (shared by all 4 M-subtiles): lane = column ncol,
        // VGPR0-3 = K pairs {0..3}+hi*4, VGPR4-7 = pairs {8..11}+hi*4
        v8u bu;
        #pragma unroll
        for (int r = 0; r < 4; ++r) {
            bu[r]     = Bs[(hi * 4 + r) * 132 + ncol];
            bu[4 + r] = Bs[(8 + hi * 4 + r) * 132 + ncol];
        }
        v16bf b = __builtin_bit_cast(v16bf, bu);

        #pragma unroll
        for (int s = 0; s < 4; ++s) {     // 4 M-subtiles -> 4 WMMAs per K-step
            const uint32_t* Arow = &As[(s * 16 + m) * 17];
            v8u au;
            #pragma unroll
            for (int r = 0; r < 4; ++r) {
                au[r]     = Arow[hi * 4 + r];
                au[4 + r] = Arow[8 + hi * 4 + r];
            }
            v16bf a = __builtin_bit_cast(v16bf, au);
            v8f* accp = (s == 0) ? &acc0 : (s == 1) ? &acc1 : (s == 2) ? &acc2 : &acc3;
            *accp = __builtin_amdgcn_wmma_f32_16x16x32_bf16(false, a, false, b,
                                                            (short)0, *accp, false, false);
        }
    }

    // C/D layout: VGPR r -> M = r (lanes 0-15) or r+8 (lanes 16-31), N = ncol
    #pragma unroll
    for (int s = 0; s < 4; ++s) {
        v8f acc = (s == 0) ? acc0 : (s == 1) ? acc1 : (s == 2) ? acc2 : acc3;
        #pragma unroll
        for (int r = 0; r < 8; ++r) {
            int om = s * 16 + r + hi * 8;
            atomicAdd(&hacc[(row0 + om) * DIM + ncol], acc[r]);
        }
    }
}

// ---------------- kernel 2: h = relu(hacc + b1) in place ----------------
__global__ void bias_relu_kernel(float* __restrict__ h, const float* __restrict__ b1) {
    int i = blockIdx.x * blockDim.x + threadIdx.x;
    if (i < NTOT * DIM) h[i] = fmaxf(h[i] + b1[i & (DIM - 1)], 0.0f);
}

// ---------------- kernel 3: E = h @ W2 + b2 (bf16 WMMA, K=128) ----------------
// grid = 20 M-tiles of 16 rows; block = 256
__global__ __launch_bounds__(256)
void gemm2_kernel(const float* __restrict__ H, const float* __restrict__ W2,
                  const float* __restrict__ b2, float* __restrict__ E)
{
    __shared__ uint32_t As[16 * 17];
    __shared__ uint32_t Bs[16 * 132];

    const int tid  = threadIdx.x;
    const int lane = tid & 31;
    const int wave = tid >> 5;
    const int row0 = blockIdx.x * 16;
    const float* X = H + (size_t)row0 * DIM;

    v8f acc = {0.f,0.f,0.f,0.f,0.f,0.f,0.f,0.f};
    const int m    = lane & 15;
    const int hi   = lane >> 4;
    const int ncol = wave * 16 + m;

    #pragma unroll
    for (int kb = 0; kb < DIM / 32; ++kb) {
        const int k0 = kb * 32;
        __syncthreads();
        {
            int am = tid >> 4, ap = tid & 15;
            const float* s = X + (size_t)am * DIM + k0 + 2 * ap;
            As[am * 17 + ap] = packbf2(s[0], s[1]);
        }
        #pragma unroll
        for (int i = 0; i < 8; ++i) {
            int idx = tid + i * 256;
            int bp = idx >> 7, bn = idx & 127;
            const float* s = W2 + (size_t)(k0 + 2 * bp) * DIM + bn;
            Bs[bp * 132 + bn] = packbf2(s[0], s[DIM]);
        }
        __syncthreads();

        v8u au, bu;
        #pragma unroll
        for (int r = 0; r < 4; ++r) {
            au[r]     = As[m * 17 + hi * 4 + r];
            au[4 + r] = As[m * 17 + 8 + hi * 4 + r];
            bu[r]     = Bs[(hi * 4 + r) * 132 + ncol];
            bu[4 + r] = Bs[(8 + hi * 4 + r) * 132 + ncol];
        }
        v16bf a = __builtin_bit_cast(v16bf, au);
        v16bf b = __builtin_bit_cast(v16bf, bu);
        acc = __builtin_amdgcn_wmma_f32_16x16x32_bf16(false, a, false, b,
                                                      (short)0, acc, false, false);
    }

    #pragma unroll
    for (int r = 0; r < 8; ++r) {
        int om = r + hi * 8;
        E[(row0 + om) * DIM + ncol] = acc[r] + b2[ncol];
    }
}

// ---------------- block reductions (128 threads = 4 wave32 waves) ----------------
__device__ __forceinline__ float block_reduce_sum(float v, float* red) {
    #pragma unroll
    for (int o = 16; o > 0; o >>= 1) v += __shfl_down(v, o, 32);
    __syncthreads();                     // protect red[] from previous use
    if ((threadIdx.x & 31) == 0) red[threadIdx.x >> 5] = v;
    __syncthreads();
    return red[0] + red[1] + red[2] + red[3];
}
__device__ __forceinline__ float block_reduce_max(float v, float* red) {
    #pragma unroll
    for (int o = 16; o > 0; o >>= 1) v = fmaxf(v, __shfl_down(v, o, 32));
    __syncthreads();
    if ((threadIdx.x & 31) == 0) red[threadIdx.x >> 5] = v;
    __syncthreads();
    return fmaxf(fmaxf(red[0], red[1]), fmaxf(red[2], red[3]));
}

// ---------------- kernel 4: ragged supervised-contrastive loss ----------------
// grid = NC blocks (one per closest-pair anchor); block = 128 (one thread per dim)
__global__ __launch_bounds__(128)
void loss_kernel(const int* __restrict__ label, const int* __restrict__ cf,
                 const int* __restrict__ iff, const int* __restrict__ cc,
                 const int* __restrict__ ic, const float* __restrict__ E,
                 float* __restrict__ lsum, float* __restrict__ lcnt)
{
    __shared__ float red[4];
    __shared__ float Avals[NF + NC];

    const int i = blockIdx.x;            // anchor index in closest pairs
    const int d = threadIdx.x;           // embedding dim
    const float* Ef = E;                 // rows [0, NF)
    const float* Ec = E + NF * DIM;      // rows [NF, NTOT)

    const int lc = label[ic[i]];
    const int c0 = cc[2 * i], c1 = cc[2 * i + 1];
    const bool isWrong = (c0 != lc) && (c1 == lc);
    const bool isCorr  = (c0 == lc);
    if (!isWrong && !isCorr) return;     // uniform across block

    const int posClass = isWrong ? c1 : c0;  // wrong: pos_of[top2]; corr: pos_of[cc0]
    const int negClass = isWrong ? c0 : c1;  // wrong: pos_of[top1]; corr: pos_of[cc1]

    // anchor = normalize(Ec[i], axis=-1)
    float av  = Ec[i * DIM + d];
    float n2  = block_reduce_sum(av * av, red);
    float anv = av / fmaxf(sqrtf(n2), EPSF);

    // positive group: column stats over members (normalize over group dim, as in ref)
    float s1 = 0.f, s2 = 0.f; int P = 0;
    for (int j = 0; j < NF; ++j) {
        int cj = cf[2 * j];
        if (cj == posClass && cj == label[iff[j]]) {
            float v = Ef[j * DIM + d];
            s1 += v; s2 += v * v; ++P;
        }
    }
    // negative group column norms: Ef part (+ Ec "neg_extra" part for wrong anchors)
    float t2 = 0.f; int Q = 0;
    for (int j = 0; j < NF; ++j) {
        int cj = cf[2 * j];
        if (cj == negClass && cj == label[iff[j]]) {
            float v = Ef[j * DIM + d];
            t2 += v * v; ++Q;
        }
    }
    if (isWrong) {
        for (int j = 0; j < NC; ++j) {
            int l2 = label[ic[j]];
            if (cc[2 * j] != l2 && cc[2 * j + 1] == l2 && cc[2 * j] == c1) {
                float v = Ec[j * DIM + d];
                t2 += v * v; ++Q;
            }
        }
    }
    if (P == 0 || Q == 0) return;        // uniform

    const float invColP = 1.f / fmaxf(sqrtf(s2), EPSF);
    const float invColN = 1.f / fmaxf(sqrtf(t2), EPSF);

    // mean over positives of (posn . an)/T, collapsed over p first
    float mean_num = block_reduce_sum(anv * s1 * invColP, red) / (TEMP * (float)P);

    // per-negative logits A[q] = (negn[q] . an)/T
    int q = 0;
    for (int j = 0; j < NF; ++j) {
        int cj = cf[2 * j];
        bool mem = (cj == negClass && cj == label[iff[j]]);   // uniform
        if (!mem) continue;
        float A = block_reduce_sum(Ef[j * DIM + d] * invColN * anv, red) / TEMP;
        if (d == 0) Avals[q] = A;
        ++q;
    }
    if (isWrong) {
        for (int j = 0; j < NC; ++j) {
            int l2 = label[ic[j]];
            bool mem = (cc[2 * j] != l2 && cc[2 * j + 1] == l2 && cc[2 * j] == c1);
            if (!mem) continue;
            float A = block_reduce_sum(Ec[j * DIM + d] * invColN * anv, red) / TEMP;
            if (d == 0) Avals[q] = A;
            ++q;
        }
    }
    __syncthreads();

    // log(sum(exp(A - max))) — max NOT added back (faithful to reference)
    float lm = -INFINITY;
    for (int k = d; k < Q; k += 128) lm = fmaxf(lm, Avals[k]);
    float mx = block_reduce_max(lm, red);
    float se = 0.f;
    for (int k = d; k < Q; k += 128) se += expf(Avals[k] - mx);
    float log_sum = logf(block_reduce_sum(se, red));

    if (d == 0) {
        float loss = -TEMP * (mean_num - log_sum);  // BASE_TEMPERATURE = 1
        atomicAdd(lsum, loss);
        atomicAdd(lcnt, 1.0f);
    }
}

// ---------------- kernel 5: finalize ----------------
__global__ void finalize_kernel(const float* __restrict__ accum, float* __restrict__ out) {
    if (threadIdx.x == 0)
        out[0] = (accum[1] > 0.0f) ? (accum[0] / accum[1]) : 0.0f;
}

// ---------------- host launcher ----------------
extern "C" void kernel_launch(void* const* d_in, const int* in_sizes, int n_in,
                              void* d_out, int out_size, void* d_ws, size_t ws_size,
                              hipStream_t stream) {
    (void)in_sizes; (void)n_in; (void)out_size; (void)ws_size;
    const int*   label = (const int*)  d_in[0];
    const float* Xf    = (const float*)d_in[1];
    const int*   cf    = (const int*)  d_in[2];
    const int*   iff   = (const int*)  d_in[3];
    const float* Xc    = (const float*)d_in[4];
    const int*   cc    = (const int*)  d_in[5];
    const int*   ic    = (const int*)  d_in[6];
    const float* W1    = (const float*)d_in[7];
    const float* b1    = (const float*)d_in[8];
    const float* W2    = (const float*)d_in[9];
    const float* b2    = (const float*)d_in[10];
    float* out = (float*)d_out;

    float* ws    = (float*)d_ws;
    float* hacc  = ws;                    // [NTOT*DIM] fp32 accumulator, then relu'd h
    float* Ebuf  = ws + NTOT * DIM;       // [NTOT*DIM] final embeddings
    float* accum = ws + 2 * NTOT * DIM;   // [2] loss sum, count

    zero_ws_kernel<<<(NTOT * DIM + 255) / 256, 256, 0, stream>>>(hacc, accum);
    gemm1_kernel<<<dim3(NTOT / MROWS, KSPLIT), 256, 0, stream>>>(Xf, Xc, W1, hacc);
    bias_relu_kernel<<<(NTOT * DIM + 255) / 256, 256, 0, stream>>>(hacc, b1);
    gemm2_kernel<<<NTOT / 16, 256, 0, stream>>>(hacc, W2, b2, Ebuf);
    loss_kernel<<<NC, 128, 0, stream>>>(label, cf, iff, cc, ic, Ebuf,
                                        &accum[0], &accum[1]);
    finalize_kernel<<<1, 32, 0, stream>>>(accum, out);
}
